// DeBERTaBlock_26620207300902
// MI455X (gfx1250) — matrix operand order repaired and checked by reading, not compiled, not run on hardware
//
#include <hip/hip_runtime.h>

#define DEVINL __device__ __forceinline__

typedef __bf16 bf16;
typedef __attribute__((ext_vector_type(16))) __bf16 v16bf;
typedef __attribute__((ext_vector_type(8)))  float  v8f;
typedef __attribute__((ext_vector_type(4)))  unsigned int u32x4;
typedef __attribute__((ext_vector_type(8)))  unsigned int u32x8;

constexpr int Bv = 2, Sv = 2048, HIDv = 1024, NHv = 16, FFv = 4096, Kclip = 512, DHv = 64;
constexpr int TWOK = 2 * Kclip;   // 1024

union FragU { v16bf v; u32x4 q[2]; };

// ---- WMMA wrapper: D = A(16x32 bf16) * B(32x16 bf16) + C(16x16 f32) ----
DEVINL v8f wmma_bf16(v16bf a, v16bf b, v8f c) {
    return __builtin_amdgcn_wmma_f32_16x16x32_bf16(
        /*neg_a=*/false, a, /*neg_b=*/false, b,
        /*c_mod=*/(short)0, c, /*reuse_a=*/false, /*reuse_b=*/false);
}

// A-fragment (16x32, M x K), A row-major, k contiguous.
// Layout: lanes 0-15 hold K=0..15, lanes 16-31 hold K=16..31 (matches frag_b), row m = lane&15.
DEVINL v16bf frag_a(const bf16* p, int ld) {
    int lane = threadIdx.x & 31;
    int hi = lane >> 4, m = lane & 15;
    const bf16* r = p + (size_t)m * ld + hi * 16;
    FragU u;
    u.q[0] = *(const u32x4*)(r);
    u.q[1] = *(const u32x4*)(r + 8);
    return u.v;
}

// B-fragment (32x16, K x N) loaded from BT[n][k] (k contiguous).
// Layout (same contiguous k order as frag_a): lanes 0-15 hold K=0..15, lanes 16-31 hold K=16..31, col n = lane&15.
DEVINL v16bf frag_b(const bf16* pT, int ld) {
    int lane = threadIdx.x & 31;
    int hi = lane >> 4, n = lane & 15;
    const bf16* r = pT + (size_t)n * ld + hi * 16;
    FragU u;
    u.q[0] = *(const u32x4*)(r);
    u.q[1] = *(const u32x4*)(r + 8);
    return u.v;
}

// B-fragment where per-lane row index is (rowBase + n) clamped to [0, TWOK-1]
// (used to read pos_k / pos_q relative-embedding rows; clamp == rel-bucket clip).
DEVINL v16bf frag_b_clamp(const bf16* base, int ld, int rowBase, int colBase) {
    int lane = threadIdx.x & 31;
    int hi = lane >> 4, n = lane & 15;
    int row = rowBase + n;
    row = row < 0 ? 0 : (row > TWOK - 1 ? TWOK - 1 : row);
    const bf16* r = base + (size_t)row * ld + colBase + hi * 16;
    FragU u;
    u.q[0] = *(const u32x4*)(r);
    u.q[1] = *(const u32x4*)(r + 8);
    return u.v;
}

// ---- Tensor Data Mover: 2-D tile (128 rows x 32 bf16) global -> LDS ----
// D# group0: [1:0]=count=1, [63:32]=lds_addr, [120:64]=global_addr, [127:126]=type(2).
// D# group1: data_size=2B, tensor_dim0=kd, tensor_dim1=128, tile=32x128, dim0_stride=kd.
DEVINL void tdm_load_tile(const bf16* g, unsigned ldsOff, int kd) {
    unsigned long long ga = (unsigned long long)(size_t)g;
    u32x4 g0;
    u32x8 g1;
    g0[0] = 1u;                                             // count=1, user mode
    g0[1] = ldsOff;                                         // lds_addr (bytes)
    g0[2] = (unsigned)ga;                                   // global_addr[31:0]
    g0[3] = ((unsigned)(ga >> 32) & 0x01FFFFFFu) | (2u << 30); // addr[56:32], type=2
    g1[0] = 1u << 16;                                       // data_size=1 -> 2 bytes
    g1[1] = ((unsigned)kd & 0xFFFFu) << 16;                 // tensor_dim0[15:0]
    g1[2] = (((unsigned)kd >> 16) & 0xFFFFu) | (128u << 16); // tensor_dim0[31:16], tensor_dim1=128
    g1[3] = (32u << 16);                                    // tile_dim0=32
    g1[4] = 128u;                                           // tile_dim1=128, tile_dim2=0
    g1[5] = (unsigned)kd;                                   // tensor_dim0_stride[31:0]
    g1[6] = 0u;
    g1[7] = 0u;
    asm volatile("tensor_load_to_lds %0, %1" :: "s"(g0), "s"(g1) : "memory");
}

// ---------------- weight convert: W[K,N] f32 -> WT[N,K] bf16 ----------------
__global__ __launch_bounds__(256) void wt_kernel(const float* __restrict__ W,
                                                 bf16* __restrict__ WT,
                                                 int Kd, int N) {
    size_t idx = (size_t)blockIdx.x * 256 + threadIdx.x;
    if (idx >= (size_t)N * Kd) return;
    int k = (int)(idx % Kd);
    size_t n = idx / Kd;
    WT[idx] = (bf16)W[(size_t)k * N + n];
}

// ---------------- plain f32 -> bf16 convert ----------------
__global__ __launch_bounds__(256) void cvt_kernel(const float* __restrict__ in,
                                                  bf16* __restrict__ o, size_t n) {
    size_t idx = (size_t)blockIdx.x * 256 + threadIdx.x;
    if (idx < n) o[idx] = (bf16)in[idx];
}

// ---------------- per-head V transpose: [B*S, HID] -> [B,NH,DH,S] ----------------
__global__ __launch_bounds__(256) void vt_kernel(const bf16* __restrict__ v,
                                                 bf16* __restrict__ vT) {
    size_t idx = (size_t)blockIdx.x * 256 + threadIdx.x;
    int s = (int)(idx % Sv);
    size_t rest = idx / Sv;
    int d = (int)(rest % DHv);
    int bh = (int)(rest / DHv);
    int b = bh / NHv, h = bh % NHv;
    vT[idx] = v[((size_t)(b * Sv + s)) * HIDv + h * DHv + d];
}

// ---------------- LayerNorm (f32 in, bf16 out), 1 row / block ----------------
__global__ __launch_bounds__(256) void ln_kernel(const float* __restrict__ x,
                                                 const float* __restrict__ g,
                                                 const float* __restrict__ bta,
                                                 bf16* __restrict__ o) {
    __shared__ float red[256];
    int row = blockIdx.x, tid = threadIdx.x;
    const float* xr = x + (size_t)row * HIDv;
    float v[4]; float s = 0.f;
#pragma unroll
    for (int i = 0; i < 4; ++i) { v[i] = xr[tid + 256 * i]; s += v[i]; }
    red[tid] = s; __syncthreads();
    for (int st = 128; st > 0; st >>= 1) { if (tid < st) red[tid] += red[tid + st]; __syncthreads(); }
    float mean = red[0] * (1.0f / HIDv);
    __syncthreads();
    float s2 = 0.f;
#pragma unroll
    for (int i = 0; i < 4; ++i) { float d = v[i] - mean; s2 += d * d; }
    red[tid] = s2; __syncthreads();
    for (int st = 128; st > 0; st >>= 1) { if (tid < st) red[tid] += red[tid + st]; __syncthreads(); }
    float rstd = rsqrtf(red[0] * (1.0f / HIDv) + 1e-6f);
#pragma unroll
    for (int i = 0; i < 4; ++i) {
        int c = tid + 256 * i;
        o[(size_t)row * HIDv + c] = (bf16)((v[i] - mean) * rstd * g[c] + bta[c]);
    }
}

// ---------------- TDM-fed, double-buffered WMMA GEMM ----------------
// C[M,N] = A[M,K] * BT[N,K]^T + bias.  Block tile 128x128, K-step 32.
// EPI 0: bf16 out. EPI 1: gelu(tanh) -> bf16. EPI 2: + res -> f32.
template <int EPI>
__global__ __launch_bounds__(256) void gemm_kernel(const bf16* __restrict__ A,
                                                   const bf16* __restrict__ BT,
                                                   const float* __restrict__ bias,
                                                   const float* __restrict__ res,
                                                   float* __restrict__ outF,
                                                   bf16* __restrict__ outB,
                                                   int M, int N, int Kd) {
    __shared__ __align__(16) bf16 sA[2][128 * 32];   // 2 x 8KB
    __shared__ __align__(16) bf16 sB[2][128 * 32];   // 2 x 8KB
    int w = threadIdx.x >> 5, lane = threadIdx.x & 31;
    int hi = lane >> 4, l15 = lane & 15;
    int m0b = blockIdx.y * 128, n0b = blockIdx.x * 128;
    int wm = (w >> 1) * 32, wn = (w & 1) * 64;

    v8f acc[2][4];
#pragma unroll
    for (int i = 0; i < 2; ++i)
#pragma unroll
        for (int t = 0; t < 4; ++t)
#pragma unroll
            for (int r = 0; r < 8; ++r) acc[i][t][r] = 0.f;

    // prologue: stage k=0 tiles
    if (threadIdx.x < 32) {
        tdm_load_tile(A + (size_t)m0b * Kd, (unsigned)(size_t)&sA[0][0], Kd);
        tdm_load_tile(BT + (size_t)n0b * Kd, (unsigned)(size_t)&sB[0][0], Kd);
    }

    for (int kk = 0; kk < Kd; kk += 32) {
        int cur = (kk >> 5) & 1;
        if (threadIdx.x < 32) {
            if (kk + 32 < Kd) {
                tdm_load_tile(A + (size_t)m0b * Kd + kk + 32,
                              (unsigned)(size_t)&sA[cur ^ 1][0], Kd);
                tdm_load_tile(BT + (size_t)n0b * Kd + kk + 32,
                              (unsigned)(size_t)&sB[cur ^ 1][0], Kd);
                // TENSORcnt is in-order: <=2 outstanding => current 2 tiles landed
                __builtin_amdgcn_s_wait_tensorcnt(2);
            } else {
                __builtin_amdgcn_s_wait_tensorcnt(0);
            }
        }
        __syncthreads();

        v16bf a0 = frag_a(&sA[cur][wm * 32], 32);
        v16bf a1 = frag_a(&sA[cur][(wm + 16) * 32], 32);
#pragma unroll
        for (int t = 0; t < 4; ++t) {
            v16bf bt = frag_b(&sB[cur][(wn + 16 * t) * 32], 32);
            acc[0][t] = wmma_bf16(a0, bt, acc[0][t]);
            acc[1][t] = wmma_bf16(a1, bt, acc[1][t]);
        }
        __syncthreads();   // protect buffer reuse against next TDM overwrite
    }

#pragma unroll
    for (int i = 0; i < 2; ++i)
#pragma unroll
        for (int t = 0; t < 4; ++t)
#pragma unroll
            for (int r = 0; r < 8; ++r) {
                int m = m0b + wm + 16 * i + hi * 8 + r;
                int n = n0b + wn + 16 * t + l15;
                float vv = acc[i][t][r] + bias[n];
                size_t idx = (size_t)m * N + n;
                if (EPI == 0) {
                    outB[idx] = (bf16)vv;
                } else if (EPI == 1) {
                    float u = 0.7978845608028654f * (vv + 0.044715f * vv * vv * vv);
                    outB[idx] = (bf16)(0.5f * vv * (1.0f + tanhf(u)));
                } else {
                    outF[idx] = vv + res[idx];
                }
            }
}

// ---------------- fused disentangled flash attention ----------------
// grid (S/64, B*NH), block 128 (4 waves); wave = one 16-row query tile.
__global__ __launch_bounds__(128) void attn_kernel(const bf16* __restrict__ q,
                                                   const bf16* __restrict__ k,
                                                   const bf16* __restrict__ vT,
                                                   const bf16* __restrict__ posk,
                                                   const bf16* __restrict__ posq,
                                                   const int* __restrict__ mask,
                                                   bf16* __restrict__ attnO) {
    __shared__ __align__(16) float ldsG[4][16 * 64];   // c2p window per wave
    __shared__ __align__(16) float ldsH[4][32 * 64];   // p2c window per wave
    __shared__ __align__(16) bf16  ldsP[4][16 * 32];   // P relayout C->A frag

    int w = threadIdx.x >> 5, lane = threadIdx.x & 31;
    int hi = lane >> 4, l15 = lane & 15;
    int bh = blockIdx.y;
    int b = bh >> 4, h = bh & 15;           // NH == 16
    int i0 = (blockIdx.x * 4 + w) * 16;

    const float sc = 0.07216878364870323f;  // 1/sqrt(3*DH)

    const bf16* qrow = q + ((size_t)(b * Sv + i0)) * HIDv + h * DHv;
    v16bf aq0 = frag_a(qrow, HIDv);         // d = 0..31
    v16bf aq1 = frag_a(qrow + 32, HIDv);    // d = 32..63

    v8f acc[4];
    float rowm[8], rowl[8];
#pragma unroll
    for (int t = 0; t < 4; ++t)
#pragma unroll
        for (int r = 0; r < 8; ++r) acc[t][r] = 0.f;
#pragma unroll
    for (int r = 0; r < 8; ++r) { rowm[r] = -1e30f; rowl[r] = 0.f; }

    for (int j0 = 0; j0 < Sv; j0 += 32) {
        // ---- c2c: Q(16x64) x K^T -> s0 (cols j0..15), s1 (cols j0+16..31) ----
        v8f s0, s1;
#pragma unroll
        for (int r = 0; r < 8; ++r) { s0[r] = 0.f; s1[r] = 0.f; }
        {
            const bf16* kr = k + ((size_t)(b * Sv + j0)) * HIDv + h * DHv;
            s0 = wmma_bf16(aq0, frag_b(kr, HIDv), s0);
            s0 = wmma_bf16(aq1, frag_b(kr + 32, HIDv), s0);
            s1 = wmma_bf16(aq0, frag_b(kr + (size_t)16 * HIDv, HIDv), s1);
            s1 = wmma_bf16(aq1, frag_b(kr + (size_t)16 * HIDv + 32, HIDv), s1);
        }

        // ---- c2p window G[ii, rr] = q_i . pos_k[clamp(r0+rr)], rr in [0,64) ----
        int r0 = i0 - j0 - 31 + Kclip;
#pragma unroll
        for (int t = 0; t < 4; ++t) {
            v8f g;
#pragma unroll
            for (int r = 0; r < 8; ++r) g[r] = 0.f;
            g = wmma_bf16(aq0, frag_b_clamp(posk, HIDv, r0 + 16 * t, h * DHv), g);
            g = wmma_bf16(aq1, frag_b_clamp(posk, HIDv, r0 + 16 * t, h * DHv + 32), g);
#pragma unroll
            for (int r = 0; r < 8; ++r)
                ldsG[w][(hi * 8 + r) * 64 + 16 * t + l15] = g[r];
        }

        // ---- p2c window H[jj, rr] = k_j . pos_q[clamp(r0p+rr)], jj in [0,32) ----
        int r0p = j0 - i0 - 15 + Kclip;
#pragma unroll
        for (int mt = 0; mt < 2; ++mt) {
            const bf16* krow = k + ((size_t)(b * Sv + j0 + 16 * mt)) * HIDv + h * DHv;
            v16bf ak0 = frag_a(krow, HIDv);
            v16bf ak1 = frag_a(krow + 32, HIDv);
#pragma unroll
            for (int t = 0; t < 4; ++t) {
                v8f hh;
#pragma unroll
                for (int r = 0; r < 8; ++r) hh[r] = 0.f;
                hh = wmma_bf16(ak0, frag_b_clamp(posq, HIDv, r0p + 16 * t, h * DHv), hh);
                hh = wmma_bf16(ak1, frag_b_clamp(posq, HIDv, r0p + 16 * t, h * DHv + 32), hh);
#pragma unroll
                for (int r = 0; r < 8; ++r)
                    ldsH[w][(16 * mt + hi * 8 + r) * 64 + 16 * t + l15] = hh[r];
            }
        }

        // ---- gather + scale + mask ----
        float p0[8], p1[8];
#pragma unroll
        for (int t2 = 0; t2 < 2; ++t2) {
            int jj = 16 * t2 + l15;
            int mv = mask[(size_t)b * Sv + j0 + jj];
#pragma unroll
            for (int r = 0; r < 8; ++r) {
                int mm = hi * 8 + r;
                float val = t2 ? s1[r] : s0[r];
                val += ldsG[w][mm * 64 + (mm - jj + 31)];
                val += ldsH[w][jj * 64 + (jj - mm + 15)];
                val *= sc;
                if (mv == 0) val = -1e9f;
                if (t2) p1[r] = val; else p0[r] = val;
            }
        }

        // ---- online softmax (row groups are 16 lanes sharing hi) ----
#pragma unroll
        for (int r = 0; r < 8; ++r) {
            float bm = fmaxf(p0[r], p1[r]);
#pragma unroll
            for (int off = 1; off < 16; off <<= 1) bm = fmaxf(bm, __shfl_xor(bm, off, 32));
            float nm = fmaxf(rowm[r], bm);
            float alpha = __expf(rowm[r] - nm);
            float e0 = __expf(p0[r] - nm), e1 = __expf(p1[r] - nm);
            p0[r] = e0; p1[r] = e1;
            float bs = e0 + e1;
#pragma unroll
            for (int off = 1; off < 16; off <<= 1) bs += __shfl_xor(bs, off, 32);
            rowl[r] = rowl[r] * alpha + bs;
            rowm[r] = nm;
#pragma unroll
            for (int t = 0; t < 4; ++t) acc[t][r] *= alpha;
        }

        // ---- P (C-layout) -> LDS -> A-fragment ----
#pragma unroll
        for (int r = 0; r < 8; ++r) {
            int mm = hi * 8 + r;
            ldsP[w][mm * 32 + l15] = (bf16)p0[r];
            ldsP[w][mm * 32 + 16 + l15] = (bf16)p1[r];
        }
        v16bf ap = frag_a(&ldsP[w][0], 32);

        // ---- O += P(16x32) x V(32x64) ----
        const bf16* vtb = vT + ((size_t)(bh * DHv)) * Sv + j0;
#pragma unroll
        for (int t = 0; t < 4; ++t) {
            v16bf bvf = frag_b(vtb + (size_t)(16 * t) * Sv, Sv);
            acc[t] = wmma_bf16(ap, bvf, acc[t]);
        }
    }

    // ---- normalize + store bf16 ----
#pragma unroll
    for (int t = 0; t < 4; ++t)
#pragma unroll
        for (int r = 0; r < 8; ++r) {
            int mm = hi * 8 + r;
            float o = acc[t][r] / rowl[r];
            attnO[((size_t)(b * Sv + i0 + mm)) * HIDv + h * DHv + 16 * t + l15] = (bf16)o;
        }
}

// =======================================================================
extern "C" void kernel_launch(void* const* d_in, const int* in_sizes, int n_in,
                              void* d_out, int out_size, void* d_ws, size_t ws_size,
                              hipStream_t stream) {
    (void)in_sizes; (void)n_in; (void)out_size; (void)ws_size;
    const float* x    = (const float*)d_in[0];
    const int*   mask = (const int*)d_in[1];
    const float* Wq = (const float*)d_in[2];  const float* bq = (const float*)d_in[3];
    const float* Wk = (const float*)d_in[4];  const float* bk = (const float*)d_in[5];
    const float* Wv = (const float*)d_in[6];  const float* bv = (const float*)d_in[7];
    const float* Wo = (const float*)d_in[8];  const float* bo = (const float*)d_in[9];
    const float* rel = (const float*)d_in[10];
    const float* Wpk = (const float*)d_in[11]; const float* bpk = (const float*)d_in[12];
    const float* Wpq = (const float*)d_in[13]; const float* bpq = (const float*)d_in[14];
    const float* ln1g = (const float*)d_in[15]; const float* ln1b = (const float*)d_in[16];
    const float* ln2g = (const float*)d_in[17]; const float* ln2b = (const float*)d_in[18];
    const float* W1 = (const float*)d_in[19]; const float* b1 = (const float*)d_in[20];
    const float* W2 = (const float*)d_in[21]; const float* b2 = (const float*)d_in[22];
    float* out = (float*)d_out;

    char* wsp = (char*)d_ws;
    auto carve = [&](size_t bytes) -> void* {
        void* p = (void*)wsp;
        wsp += (bytes + 255) & ~(size_t)255;
        return p;
    };
    const size_t HH = (size_t)HIDv * HIDv;
    const size_t MH = (size_t)Bv * Sv * HIDv;       // 4096 x 1024
    bf16* WqT  = (bf16*)carve(HH * 2);
    bf16* WkT  = (bf16*)carve(HH * 2);
    bf16* WvT  = (bf16*)carve(HH * 2);
    bf16* WoT  = (bf16*)carve(HH * 2);
    bf16* W1T  = (bf16*)carve((size_t)HIDv * FFv * 2);
    bf16* W2T  = (bf16*)carve((size_t)HIDv * FFv * 2);
    bf16* WpkT = (bf16*)carve(HH * 2);
    bf16* WpqT = (bf16*)carve(HH * 2);
    bf16* relb = (bf16*)carve((size_t)TWOK * HIDv * 2);
    bf16* xn1  = (bf16*)carve(MH * 2);
    bf16* qb   = (bf16*)carve(MH * 2);
    bf16* kb   = (bf16*)carve(MH * 2);
    bf16* vb   = (bf16*)carve(MH * 2);
    bf16* vTb  = (bf16*)carve(MH * 2);
    bf16* posk = (bf16*)carve((size_t)TWOK * HIDv * 2);
    bf16* posq = (bf16*)carve((size_t)TWOK * HIDv * 2);
    bf16* attnb = (bf16*)carve(MH * 2);
    float* hbuf = (float*)carve(MH * 4);
    bf16* xn2  = (bf16*)carve(MH * 2);
    bf16* ffn1 = (bf16*)carve((size_t)Bv * Sv * FFv * 2);

    dim3 blk256(256), blk128(128);
    int rowsBS = Bv * Sv;                            // 4096

    // weights -> bf16 transposed
    wt_kernel<<<(int)(HH / 256), blk256, 0, stream>>>(Wq, WqT, HIDv, HIDv);
    wt_kernel<<<(int)(HH / 256), blk256, 0, stream>>>(Wk, WkT, HIDv, HIDv);
    wt_kernel<<<(int)(HH / 256), blk256, 0, stream>>>(Wv, WvT, HIDv, HIDv);
    wt_kernel<<<(int)(HH / 256), blk256, 0, stream>>>(Wo, WoT, HIDv, HIDv);
    wt_kernel<<<(int)((size_t)HIDv * FFv / 256), blk256, 0, stream>>>(W1, W1T, HIDv, FFv);
    wt_kernel<<<(int)((size_t)HIDv * FFv / 256), blk256, 0, stream>>>(W2, W2T, FFv, HIDv);
    wt_kernel<<<(int)(HH / 256), blk256, 0, stream>>>(Wpk, WpkT, HIDv, HIDv);
    wt_kernel<<<(int)(HH / 256), blk256, 0, stream>>>(Wpq, WpqT, HIDv, HIDv);
    cvt_kernel<<<(int)((size_t)TWOK * HIDv / 256), blk256, 0, stream>>>(rel, relb, (size_t)TWOK * HIDv);

    // LN1
    ln_kernel<<<rowsBS, blk256, 0, stream>>>(x, ln1g, ln1b, xn1);

    // Q/K/V projections (bf16 out)
    gemm_kernel<0><<<dim3(HIDv / 128, rowsBS / 128), blk256, 0, stream>>>(xn1, WqT, bq, nullptr, nullptr, qb, rowsBS, HIDv, HIDv);
    gemm_kernel<0><<<dim3(HIDv / 128, rowsBS / 128), blk256, 0, stream>>>(xn1, WkT, bk, nullptr, nullptr, kb, rowsBS, HIDv, HIDv);
    gemm_kernel<0><<<dim3(HIDv / 128, rowsBS / 128), blk256, 0, stream>>>(xn1, WvT, bv, nullptr, nullptr, vb, rowsBS, HIDv, HIDv);

    // pos_k / pos_q projections
    gemm_kernel<0><<<dim3(HIDv / 128, TWOK / 128), blk256, 0, stream>>>(relb, WpkT, bpk, nullptr, nullptr, posk, TWOK, HIDv, HIDv);
    gemm_kernel<0><<<dim3(HIDv / 128, TWOK / 128), blk256, 0, stream>>>(relb, WpqT, bpq, nullptr, nullptr, posq, TWOK, HIDv, HIDv);

    // V -> per-head transposed [B,NH,DH,S]
    vt_kernel<<<(int)(MH / 256), blk256, 0, stream>>>(vb, vTb);

    // fused disentangled flash attention
    attn_kernel<<<dim3(Sv / 64, Bv * NHv), blk128, 0, stream>>>(qb, kb, vTb, posk, posq, mask, attnb);

    // output projection + residual -> h (f32)
    gemm_kernel<2><<<dim3(HIDv / 128, rowsBS / 128), blk256, 0, stream>>>(attnb, WoT, bo, x, hbuf, nullptr, rowsBS, HIDv, HIDv);

    // LN2
    ln_kernel<<<rowsBS, blk256, 0, stream>>>(hbuf, ln2g, ln2b, xn2);

    // FFN
    gemm_kernel<1><<<dim3(FFv / 128, rowsBS / 128), blk256, 0, stream>>>(xn2, W1T, b1, nullptr, nullptr, ffn1, rowsBS, FFv, HIDv);
    gemm_kernel<2><<<dim3(HIDv / 128, rowsBS / 128), blk256, 0, stream>>>(ffn1, W2T, b2, hbuf, out, nullptr, rowsBS, HIDv, FFv);
}
